// KnowledgeCircuit_88124138979391
// MI455X (gfx1250) — matrix operand order repaired and checked by reading, not compiled, not run on hardware
//
#include <hip/hip_runtime.h>

// ---------------------------------------------------------------------------
// KnowledgeCircuit on MI455X (gfx1250, wave32, WMMA bf16 16x16x32, f32 acc)
//
// h[tok,r]   = sum_n fw[tok,n] * (X[tok,:] @ FK[n,:,:])[r]          (137 GFLOP)
// out[tok,d] = sum_{n,r} (h[tok,r]*rw[tok,n]) * RK[n,r,d]           (137 GFLOP)
//
// Compute-bound => use v_wmma_f32_16x16x32_bf16. Weights repacked once into
// WMMA B-fragment order (bf16) so the GEMM hot loops do contiguous
// global_load_b128 per fragment; A fragments come from LDS (stage1: bf16 X
// tile; stage2: generated on the fly from h*rw).
// ---------------------------------------------------------------------------

typedef __attribute__((ext_vector_type(16))) __bf16 v16bf;
typedef __attribute__((ext_vector_type(8)))  __bf16 v8bf;
typedef __attribute__((ext_vector_type(8)))  float  v8f;
typedef __attribute__((ext_vector_type(4)))  float  v4f;

#define WMMA_BF16(A, B, C) \
  __builtin_amdgcn_wmma_f32_16x16x32_bf16(false, (A), false, (B), (short)0, (C), false, false)

static constexpr int kD = 1024;   // feature dim
static constexpr int kN = 64;     // neurons
static constexpr int kR = 128;    // rank
// tokens total = B*S = 8192

// --------------------------------------------------------------------------
// Pack feature_know [N, D, R] (f32) -> bf16 WMMA B fragments.
// Fragment index: (((n*(D/32) + kc)*(R/16) + rt)*32 + lane)*16 bf16 elements.
// B layout (16-bit 32x16): N = lane%16, elem i <-> K = i + (i>=8 ? 8 : 0) + 8*(lane>=16).
// 524288 threads, each emits one 32-byte lane fragment (coalesced b128 x2).
// --------------------------------------------------------------------------
__global__ void __launch_bounds__(256) pack_fk(const float* __restrict__ fk,
                                               __bf16* __restrict__ pk) {
  const int t    = blockIdx.x * 256 + threadIdx.x;   // [0, 64*32*8*32)
  const int lane = t & 31;
  const int rt   = (t >> 5) & 7;
  const int kc   = (t >> 8) & 31;
  const int n    = t >> 13;
  const int nn   = lane & 15;
  const int hi   = lane >> 4;
  const int r    = rt * 16 + nn;
  v16bf v;
#pragma unroll
  for (int i = 0; i < 16; ++i) {
    const int kk = i + (i >= 8 ? 8 : 0) + 8 * hi;   // B-fragment K mapping
    const int d  = kc * 32 + kk;
    v[i] = (__bf16)fk[((size_t)n * kD + d) * kR + r];
  }
  *(v16bf*)(pk + (size_t)t * 16) = v;
}

// --------------------------------------------------------------------------
// Pack restore_know [N, R, D] = flat [8192, 1024] (f32) -> bf16 B fragments.
// Composite K = n*128 + r (256 chunks of 32), N-dim = d (64 tiles of 16).
// Fragment index: ((kc*(D/16) + dt)*32 + lane)*16.
// --------------------------------------------------------------------------
__global__ void __launch_bounds__(256) pack_rk(const float* __restrict__ rk,
                                               __bf16* __restrict__ pk) {
  const int t    = blockIdx.x * 256 + threadIdx.x;   // [0, 256*64*32)
  const int lane = t & 31;
  const int dt   = (t >> 5) & 63;
  const int kc   = t >> 11;
  const int nn   = lane & 15;
  const int hi   = lane >> 4;
  const int d    = dt * 16 + nn;
  v16bf v;
#pragma unroll
  for (int i = 0; i < 16; ++i) {
    const int kk = i + (i >= 8 ? 8 : 0) + 8 * hi;
    const int k  = kc * 32 + kk;           // row of flat [N*R, D]
    v[i] = (__bf16)rk[(size_t)k * kD + d];
  }
  *(v16bf*)(pk + (size_t)t * 16) = v;
}

// --------------------------------------------------------------------------
// Stage 1: h[tok, r] = sum_n fw[tok,n] * (X @ FK[n])[tok, r]
// Block = 256 threads (8 waves), 32-token tile; wave w owns r-tile w.
// X tile staged in LDS as bf16, stride 1032 (bank spread). Per n: 32 WMMAs
// over K=1024, then per-row scale by fw into running f32 accumulators.
// A layout (16-bit 16x32): M = lane%16, K = i + (i>=8 ? 8 : 0) + 8*(lane>=16)
//   => two contiguous 8-elem runs: [kc*32+8*hi, +8) and [kc*32+16+8*hi, +8).
// --------------------------------------------------------------------------
__global__ void __launch_bounds__(256) stage1(const float* __restrict__ x,
                                              const float* __restrict__ fw,
                                              const __bf16* __restrict__ fkp,
                                              float* __restrict__ h) {
  extern __shared__ char smem[];
  __bf16* Xs = (__bf16*)smem;                          // [32][1032] bf16
  float*  Ws = (float*)(smem + 32 * 1032 * 2);         // [32][64]  f32

  const int tid     = threadIdx.x;
  const int wave    = tid >> 5;
  const int lane    = tid & 31;
  const int nn      = lane & 15;
  const int hi      = lane >> 4;
  const int tokBase = blockIdx.x * 32;

  // Stage X tile (32 x 1024 f32 -> bf16), vectorized.
  for (int idx = tid; idx < 32 * 256; idx += 256) {
    const int row = idx >> 8;
    const int c4  = (idx & 255) << 2;
    const float4 val = *(const float4*)(x + (size_t)(tokBase + row) * kD + c4);
    __bf16* dst = Xs + row * 1032 + c4;
    dst[0] = (__bf16)val.x; dst[1] = (__bf16)val.y;
    dst[2] = (__bf16)val.z; dst[3] = (__bf16)val.w;
  }
  // Stage fw tile (32 x 64 f32).
  for (int idx = tid; idx < 32 * 64; idx += 256) {
    const int row = idx >> 6;
    const int c   = idx & 63;
    Ws[idx] = fw[(size_t)(tokBase + row) * kN + c];
  }
  __syncthreads();

  v8f hacc0 = {}, hacc1 = {};
  for (int n = 0; n < kN; ++n) {
    v8f acc0 = {}, acc1 = {};
    const __bf16* bp = fkp + ((((size_t)n * 32) * 8 + wave) * 32 + lane) * 16;
#pragma unroll 2
    for (int kc = 0; kc < 32; ++kc) {
      const v16bf b = *(const v16bf*)bp;
      bp += 8 * 32 * 16;                                 // next kc chunk
      const int dbase = kc * 32 + 8 * hi;
      const v8bf a0lo = *(const v8bf*)(Xs + nn * 1032 + dbase);
      const v8bf a0hi = *(const v8bf*)(Xs + nn * 1032 + dbase + 16);
      const v16bf a0  = __builtin_shufflevector(a0lo, a0hi,
          0,1,2,3,4,5,6,7,8,9,10,11,12,13,14,15);
      acc0 = WMMA_BF16(a0, b, acc0);
      const v8bf a1lo = *(const v8bf*)(Xs + (16 + nn) * 1032 + dbase);
      const v8bf a1hi = *(const v8bf*)(Xs + (16 + nn) * 1032 + dbase + 16);
      const v16bf a1  = __builtin_shufflevector(a1lo, a1hi,
          0,1,2,3,4,5,6,7,8,9,10,11,12,13,14,15);
      acc1 = WMMA_BF16(a1, b, acc1);
    }
    // C layout: elem v <-> (M = v + 8*hi, N = nn); scale per-row by fw.
#pragma unroll
    for (int v = 0; v < 8; ++v) {
      hacc0[v] += Ws[(v + 8 * hi) * 64 + n] * acc0[v];
      hacc1[v] += Ws[(16 + v + 8 * hi) * 64 + n] * acc1[v];
    }
  }
#pragma unroll
  for (int v = 0; v < 8; ++v) {
    h[(size_t)(tokBase + v + 8 * hi) * kR + wave * 16 + nn]      = hacc0[v];
    h[(size_t)(tokBase + 16 + v + 8 * hi) * kR + wave * 16 + nn] = hacc1[v];
  }
}

// --------------------------------------------------------------------------
// Stage 2: out[tok, d] = sum_{k=(n,r)} (h[tok,r]*rw[tok,n]) * RK[k, d]
// Block = 256 threads (8 waves), 16-token tile; wave w owns d-tiles 8w..8w+7.
// A fragment generated on the fly from LDS (h*rw, f32->bf16) and reused
// across 8 WMMAs (one per d-tile). 256 K-chunks of 32 (each within one n).
// --------------------------------------------------------------------------
__global__ void __launch_bounds__(256) stage2(const float* __restrict__ h,
                                              const float* __restrict__ rw,
                                              const __bf16* __restrict__ rkp,
                                              float* __restrict__ out) {
  __shared__ __attribute__((aligned(16))) float Hs[16 * 132];  // padded rows
  __shared__ __attribute__((aligned(16))) float Rw[16 * 64];

  const int tid     = threadIdx.x;
  const int wave    = tid >> 5;
  const int lane    = tid & 31;
  const int nn      = lane & 15;
  const int hi      = lane >> 4;
  const int tokBase = blockIdx.x * 16;

  for (int idx = tid; idx < 16 * 32; idx += 256) {
    const int row = idx >> 5;
    const int c4  = (idx & 31) << 2;
    *(v4f*)(Hs + row * 132 + c4) =
        *(const v4f*)(h + (size_t)(tokBase + row) * kR + c4);
  }
  for (int idx = tid; idx < 16 * 64; idx += 256) {
    const int row = idx >> 6;
    const int c   = idx & 63;
    Rw[idx] = rw[(size_t)(tokBase + row) * kN + c];
  }
  __syncthreads();

  v8f acc[8];
#pragma unroll
  for (int j = 0; j < 8; ++j) acc[j] = (v8f){};

  for (int kc = 0; kc < 256; ++kc) {
    const int n     = kc >> 2;
    const int rbase = (kc & 3) << 5;
    const float rwv = Rw[nn * 64 + n];                  // A row M = nn
    const float* hp = Hs + nn * 132 + rbase + 8 * hi;
    const v4f f0 = *(const v4f*)(hp);
    const v4f f1 = *(const v4f*)(hp + 4);
    const v4f f2 = *(const v4f*)(hp + 16);
    const v4f f3 = *(const v4f*)(hp + 20);
    v16bf a;
#pragma unroll
    for (int i = 0; i < 4; ++i) {
      a[i]      = (__bf16)(f0[i] * rwv);   // K = i      + 8*hi
      a[4 + i]  = (__bf16)(f1[i] * rwv);   // K = 4 + i  + 8*hi
      a[8 + i]  = (__bf16)(f2[i] * rwv);   // K = 16 + i + 8*hi
      a[12 + i] = (__bf16)(f3[i] * rwv);   // K = 20 + i + 8*hi
    }
    const __bf16* bp = rkp + (((size_t)kc * 64 + wave * 8) * 32 + lane) * 16;
#pragma unroll
    for (int j = 0; j < 8; ++j) {
      const v16bf b = *(const v16bf*)(bp + (size_t)j * 32 * 16);
      acc[j] = WMMA_BF16(a, b, acc[j]);
    }
  }
#pragma unroll
  for (int j = 0; j < 8; ++j)
#pragma unroll
    for (int v = 0; v < 8; ++v)
      out[(size_t)(tokBase + v + 8 * hi) * kD + (wave * 8 + j) * 16 + nn] =
          acc[j][v];
}

// --------------------------------------------------------------------------
// Launch. Inputs: x, feature_know_w, restore_know_w, feature_know,
// restore_know (all f32). Workspace: FKpack 16MB | RKpack 16MB | h 4MB.
// --------------------------------------------------------------------------
extern "C" void kernel_launch(void* const* d_in, const int* in_sizes, int n_in,
                              void* d_out, int out_size, void* d_ws,
                              size_t ws_size, hipStream_t stream) {
  (void)in_sizes; (void)n_in; (void)out_size; (void)ws_size;
  const float* x  = (const float*)d_in[0];
  const float* fw = (const float*)d_in[1];
  const float* rw = (const float*)d_in[2];
  const float* fk = (const float*)d_in[3];
  const float* rk = (const float*)d_in[4];
  float* out = (float*)d_out;

  char* ws = (char*)d_ws;
  __bf16* fkp = (__bf16*)ws;                              // 16 MB
  __bf16* rkp = (__bf16*)(ws + ((size_t)16 << 20));       // 16 MB
  float*  h   = (float*)(ws + ((size_t)32 << 20));        // 4 MB

  pack_fk<<<2048, 256, 0, stream>>>(fk, fkp);
  pack_rk<<<2048, 256, 0, stream>>>(rk, rkp);

  const size_t smem1 = (size_t)32 * 1032 * 2 + (size_t)32 * 64 * 4; // 74240 B
  stage1<<<256, 256, smem1, stream>>>(x, fw, fkp, h);
  stage2<<<512, 256, 0, stream>>>(h, rw, rkp, out);
}
